// forecastRNN_14396730376704
// MI455X (gfx1250) — matrix operand order repaired and checked by reading, not compiled, not run on hardware
//
#include <hip/hip_runtime.h>
#include <math.h>

typedef __attribute__((ext_vector_type(16))) __bf16 v16bf;
typedef __attribute__((ext_vector_type(8)))  float  v8f;
typedef __attribute__((ext_vector_type(4)))  unsigned int v4u;
typedef __attribute__((ext_vector_type(8)))  int v8i;
typedef __attribute__((ext_vector_type(4)))  int v4i;

#define NF   512
#define HIDP 1024   // HID=1000 padded to 1024
#define TT   65
#define TS   64
#define BSZ  512

// ---------------- prep kernels ----------------

// WcatT[k][n] (k<1024, n<512): k<512 -> W_ih[n][k], else W_hh[n][k-512]
__global__ void prep_wcat(const float* __restrict__ Wih, const float* __restrict__ Whh,
                          __bf16* __restrict__ WT) {
  int idx = blockIdx.x * 256 + threadIdx.x;
  if (idx >= 1024 * NF) return;
  int k = idx >> 9, n = idx & (NF - 1);
  float v = (k < NF) ? Wih[n * NF + k] : Whh[n * NF + (k - NF)];
  WT[(size_t)k * NF + n] = (__bf16)v;
}

__global__ void prep_bcat(const float* a, const float* b, float* o) {
  int i = blockIdx.x * 256 + threadIdx.x;
  if (i < NF) o[i] = a[i] + b[i];
}

// BN-folded, transposed, padded weight: WT[k][n] = W[n][k] * g[n]*rsqrt(rv[n]+eps)
// bias[n] = (b[n]-rm[n])*s + beta[n]; zero outside valid range.
__global__ void prep_scaled(const float* __restrict__ W, int Nv, int Kv, int Npad, int Kpad,
                            const float* __restrict__ g, const float* __restrict__ beta,
                            const float* __restrict__ rm, const float* __restrict__ rv,
                            const float* __restrict__ b,
                            __bf16* __restrict__ WT, float* __restrict__ bias) {
  int idx = blockIdx.x * 256 + threadIdx.x;
  if (idx >= Kpad * Npad) return;
  int k = idx / Npad, n = idx - k * Npad;
  float s = (n < Nv) ? g[n] * rsqrtf(rv[n] + 1e-5f) : 0.f;
  float v = (n < Nv && k < Kv) ? W[n * Kv + k] * s : 0.f;
  WT[idx] = (__bf16)v;
  if (k == 0) bias[n] = (n < Nv) ? (b[n] - rm[n]) * s + beta[n] : 0.f;
}

__global__ void zero_init(__bf16* __restrict__ h, int n, float* __restrict__ acc) {
  int i = blockIdx.x * blockDim.x + threadIdx.x;
  if (i < n) h[i] = (__bf16)0.f;
  if (i < 2) acc[i] = 0.f;
}

// convert one timestep slab x[:, t, :] (f32, row stride lds_) -> bf16 [BSZ][NF]
__global__ void cvt_rows(const float* __restrict__ src, int lds_, __bf16* __restrict__ dst) {
  int i = blockIdx.x * 256 + threadIdx.x;
  if (i >= BSZ * NF) return;
  int b = i >> 9, f = i & (NF - 1);
  dst[i] = (__bf16)src[(size_t)b * lds_ + f];
}

// ---------------- TDM: stage a 32(K) x 128(N) bf16 weight panel into LDS ----------------
// D# per CDNA5 ISA: group0 = {flags, lds_addr, global_addr, type=2},
// group1 = {data_size=2B, tensor_dim0=ldn, tensor_dim1=32, tile=[128,32], dim0_stride=ldn}.
// This toolchain exposes the 6-arg builtin: (v4u g0, v8i g1, v4i g2, v4i g3, v8i extra, i32 cpol).
__device__ __forceinline__ void tdm_load_panel(const __bf16* gsrc, unsigned ldsOff, int ldn) {
  unsigned long long ga = (unsigned long long)(uintptr_t)gsrc;
  v4u g0;
  g0[0] = 1u;                                   // count=1, user descriptor
  g0[1] = ldsOff;                               // LDS byte address (smem is the only LDS object)
  g0[2] = (unsigned)ga;                         // global addr [31:0]
  g0[3] = (unsigned)(ga >> 32) | (2u << 30);    // global addr [56:32] | type=2
  v8i g1;
  g1[0] = 0x00010000;                           // workgroup_mask=0, data_size=1 (2 bytes)
  g1[1] = (int)(((unsigned)ldn & 0xFFFFu) << 16);   // tensor_dim0[15:0]
  g1[2] = (int)(32u << 16);                     // tensor_dim0[31:16]=0 | tensor_dim1[15:0]=32
  g1[3] = (int)(128u << 16);                    // tensor_dim1[31:16]=0 | tile_dim0=128
  g1[4] = 32;                                   // tile_dim1=32, tile_dim2=0
  g1[5] = ldn;                                  // tensor_dim0_stride[31:0] (elements)
  g1[6] = 0;
  g1[7] = 0;
  v4i gz4 = {0, 0, 0, 0};                       // group2/group3: 2D tensor -> unused (zeros)
  v8i gz8 = {0, 0, 0, 0, 0, 0, 0, 0};
  __builtin_amdgcn_tensor_load_to_lds(g0, g1, gz4, gz4, gz8, 0);
}

// ---------------- WMMA GEMM ----------------
// 16-bit A 16x32 fragment: lane L(0..15): M=L half 0; lane L+16: same M half 1.
// element j(0..7) -> K = half*8 + j ; j(8..15) -> K = 16 + half*8 + (j-8)
__device__ __forceinline__ v16bf load_a_bf(const __bf16* __restrict__ A, int lda,
                                           int m0, int k0, int lane) {
  const __bf16* p = A + (size_t)(m0 + (lane & 15)) * lda + k0 + ((lane >> 4) << 3);
  v16bf r;
  ((uint4*)&r)[0] = *(const uint4*)p;
  ((uint4*)&r)[1] = *(const uint4*)(p + 16);
  return r;
}

// C/D 16x16 f32 layout: VGPR r, lane l -> row = r + 8*(l>>4), col = l&15
__device__ __forceinline__ float epilogue(float* __restrict__ Cf, __bf16* __restrict__ Cb,
                                          int ldc, int m0, int n0, int lane, v8f acc,
                                          const float* __restrict__ bias, int do_tanh,
                                          const float* __restrict__ lossX, int ldx) {
  int col   = n0 + (lane & 15);
  int rbase = m0 + ((lane >> 4) << 3);
  float bv = bias[col];
  float s = 0.f;
#pragma unroll
  for (int r = 0; r < 8; ++r) {
    float v = acc[r] + bv;
    if (do_tanh) v = tanhf(v);
    if (Cf) Cf[(size_t)(rbase + r) * ldc + col] = v;
    if (Cb) Cb[(size_t)(rbase + r) * ldc + col] = (__bf16)v;
    if (lossX) {
      float d = v - lossX[(size_t)(rbase + r) * ldx + col];
      s += d * d;
    }
  }
  return s;
}

// C[M,N] = act( concat(A1,A2)[M,K](bf16) @ WT[K,N](bf16) + bias )
// grid = (M/64, N/128), block = 256 (8 waves, 2x4), wave tile 32x32.
// Weight panel double-buffered in LDS via Tensor Data Mover.
__global__ void gemm_wmma(const __bf16* __restrict__ A1, int lda1,
                          const __bf16* __restrict__ A2, int lda2,
                          int Ksplit, int K,
                          const __bf16* __restrict__ WT, int ldn,
                          const float* __restrict__ bias,
                          float* __restrict__ Cf, __bf16* __restrict__ Cb, int ldc,
                          int do_tanh,
                          const float* __restrict__ lossX, int ldx,
                          float* __restrict__ lossAcc) {
  __shared__ __bf16 smem[2][32 * 128];   // only LDS object -> LDS byte offset 0
  const int lane = threadIdx.x & 31;
  const int w    = threadIdx.x >> 5;
  const int m0   = blockIdx.x * 64 + (w >> 2) * 32;
  const int nblk = blockIdx.y * 128;
  const int nw   = (w & 3) * 32;
  const int n0   = nblk + nw;
  const int nk   = K >> 5;

  v8f c00 = {}, c01 = {}, c10 = {}, c11 = {};

  if (w == 0) tdm_load_panel(WT + nblk, 0u, ldn);

  for (int kc = 0; kc < nk; ++kc) {
    if (w == 0) __builtin_amdgcn_s_wait_tensorcnt(0);
    __syncthreads();   // publish smem[kc&1]; also fences last round's reads of the other buffer
    if (w == 0 && kc + 1 < nk)
      tdm_load_panel(WT + (size_t)(kc + 1) * 32 * ldn + nblk,
                     (unsigned)(((kc + 1) & 1) * (32 * 128 * 2)), ldn);

    const int k0 = kc << 5;
    const __bf16* Ab; int lda, kk;
    if (k0 < Ksplit) { Ab = A1; lda = lda1; kk = k0; }
    else             { Ab = A2; lda = lda2; kk = k0 - Ksplit; }

    v16bf a0 = load_a_bf(Ab, lda, m0,      kk, lane);
    v16bf a1 = load_a_bf(Ab, lda, m0 + 16, kk, lane);

    // B fragment: lane = N column (lane&15); element s -> K = 8*half + s (s<8), 16 + 8*half + (s-8) (s>=8)
    const __bf16* bp = &smem[kc & 1][((lane >> 4) << 3) * 128 + nw + (lane & 15)];
    v16bf b0, b1;
#pragma unroll
    for (int s = 0; s < 16; ++s) {
      int kr = (s < 8) ? s : s + 8;
      b0[s] = bp[kr * 128];
      b1[s] = bp[kr * 128 + 16];
    }

    c00 = __builtin_amdgcn_wmma_f32_16x16x32_bf16(false, a0, false, b0, (short)0, c00, false, false);
    c01 = __builtin_amdgcn_wmma_f32_16x16x32_bf16(false, a0, false, b1, (short)0, c01, false, false);
    c10 = __builtin_amdgcn_wmma_f32_16x16x32_bf16(false, a1, false, b0, (short)0, c10, false, false);
    c11 = __builtin_amdgcn_wmma_f32_16x16x32_bf16(false, a1, false, b1, (short)0, c11, false, false);
  }

  float s = 0.f;
  s += epilogue(Cf, Cb, ldc, m0,      n0,      lane, c00, bias, do_tanh, lossX, ldx);
  s += epilogue(Cf, Cb, ldc, m0,      n0 + 16, lane, c01, bias, do_tanh, lossX, ldx);
  s += epilogue(Cf, Cb, ldc, m0 + 16, n0,      lane, c10, bias, do_tanh, lossX, ldx);
  s += epilogue(Cf, Cb, ldc, m0 + 16, n0 + 16, lane, c11, bias, do_tanh, lossX, ldx);

  if (lossX && lossAcc) {
    for (int o = 16; o; o >>= 1) s += __shfl_down(s, o, 32);
    if (lane == 0) atomicAdd(lossAcc, s);
  }
}

// ---------------- epilogue kernels ----------------

__global__ void select_pred(const float* __restrict__ y, const int* __restrict__ t,
                            int step, float* __restrict__ out) {
  int b = blockIdx.x;
  if (t[b] - 1 != step) return;
  for (int f = threadIdx.x; f < NF; f += blockDim.x)
    out[(size_t)b * NF + f] = y[(size_t)b * NF + f];
}

__global__ void loss2_kernel(const float* __restrict__ pred, const float* __restrict__ x,
                             float* __restrict__ lossAcc) {
  int b = blockIdx.x;
  float s = 0.f;
  for (int f = threadIdx.x; f < NF; f += blockDim.x) {
    float d = pred[(size_t)b * NF + f] - x[((size_t)b * TT + (TT - 1)) * NF + f];
    s += d * d;
  }
  for (int o = 16; o; o >>= 1) s += __shfl_down(s, o, 32);
  if ((threadIdx.x & 31) == 0) atomicAdd(lossAcc + 1, s);
}

__global__ void finalize(const float* __restrict__ acc, float* __restrict__ out) {
  double n1 = 63.0 * 512.0 * 512.0;   // (T-1)*B*F ; loss1 = sum/(n1*n1)
  double n2 = 512.0 * 512.0;          // B*F       ; loss2 = sum/(n2*n2)
  out[0] = (float)((double)acc[0] / (n1 * n1) + (double)acc[1] / (n2 * n2));
}

// ---------------- driver ----------------

extern "C" void kernel_launch(void* const* d_in, const int* in_sizes, int n_in,
                              void* d_out, int out_size, void* d_ws, size_t ws_size,
                              hipStream_t stream) {
  const float* x   = (const float*)d_in[0];
  const int*   tt  = (const int*)d_in[1];
  const float* Wih = (const float*)d_in[2];
  const float* Whh = (const float*)d_in[3];
  const float* bih = (const float*)d_in[4];
  const float* bhh = (const float*)d_in[5];
  const float* W1  = (const float*)d_in[6];  const float* b1  = (const float*)d_in[7];
  const float* g1  = (const float*)d_in[8];  const float* be1 = (const float*)d_in[9];
  const float* rm1 = (const float*)d_in[10]; const float* rv1 = (const float*)d_in[11];
  const float* W2  = (const float*)d_in[12]; const float* b2  = (const float*)d_in[13];
  const float* g2  = (const float*)d_in[14]; const float* be2 = (const float*)d_in[15];
  const float* rm2 = (const float*)d_in[16]; const float* rv2 = (const float*)d_in[17];
  const float* W3  = (const float*)d_in[18]; const float* b3  = (const float*)d_in[19];
  const float* g3  = (const float*)d_in[20]; const float* be3 = (const float*)d_in[21];
  const float* rm3 = (const float*)d_in[22]; const float* rv3 = (const float*)d_in[23];

  char* ws = (char*)d_ws;
  size_t off = 0;
  auto carve = [&](size_t bytes) -> void* {
    void* p = ws + off;
    off = (off + bytes + 255) & ~(size_t)255;
    return p;
  };
  __bf16* WcatT = (__bf16*)carve((size_t)1024 * NF * 2);
  __bf16* W1T   = (__bf16*)carve((size_t)NF * HIDP * 2);
  __bf16* W2T   = (__bf16*)carve((size_t)HIDP * HIDP * 2);
  __bf16* W3T   = (__bf16*)carve((size_t)HIDP * NF * 2);
  float* bcat = (float*)carve(NF * 4);
  float* bb1  = (float*)carve(HIDP * 4);
  float* bb2  = (float*)carve(HIDP * 4);
  float* bb3  = (float*)carve(NF * 4);
  __bf16* xtb = (__bf16*)carve((size_t)BSZ * NF * 2);
  __bf16* hA  = (__bf16*)carve((size_t)BSZ * NF * 2);
  __bf16* hB  = (__bf16*)carve((size_t)BSZ * NF * 2);
  __bf16* z1  = (__bf16*)carve((size_t)BSZ * HIDP * 2);
  __bf16* z2  = (__bf16*)carve((size_t)BSZ * HIDP * 2);
  float*  yb  = (float*)carve((size_t)BSZ * NF * 4);
  __bf16* ybb = (__bf16*)carve((size_t)BSZ * NF * 2);
  float*  lacc = (float*)carve(256);
  float*  outp = (float*)d_out;

  // weight prep (BN folded, bf16, [K][N] transposed, zero-padded)
  prep_wcat<<<(1024 * NF + 255) / 256, 256, 0, stream>>>(Wih, Whh, WcatT);
  prep_bcat<<<2, 256, 0, stream>>>(bih, bhh, bcat);
  prep_scaled<<<(NF * HIDP + 255) / 256, 256, 0, stream>>>(W1, 1000, NF,  HIDP, NF,  g1, be1, rm1, rv1, b1, W1T, bb1);
  prep_scaled<<<(HIDP * HIDP + 255) / 256, 256, 0, stream>>>(W2, 1000, 1000, HIDP, HIDP, g2, be2, rm2, rv2, b2, W2T, bb2);
  prep_scaled<<<(HIDP * NF + 255) / 256, 256, 0, stream>>>(W3, NF, 1000, NF, HIDP, g3, be3, rm3, rv3, b3, W3T, bb3);
  zero_init<<<(BSZ * NF + 255) / 256, 256, 0, stream>>>(hA, BSZ * NF, lacc);

  dim3 blk(256);
  dim3 gN512(BSZ / 64, NF / 128);     // M=512, N=512
  dim3 gN1024(BSZ / 64, HIDP / 128);  // M=512, N=1024

  // ---- RNN scan + fused autoencoder + loss1 ----
  for (int t = 0; t < TS; ++t) {
    const __bf16* hin  = (t & 1) ? hB : hA;
    __bf16*       hout = (t & 1) ? hA : hB;
    cvt_rows<<<(BSZ * NF + 255) / 256, 256, 0, stream>>>(x + (size_t)t * NF, TT * NF, xtb);
    // h' = tanh([x_t, h] @ WcatT + bcat)   (K = 512 + 512)
    gemm_wmma<<<gN512, blk, 0, stream>>>(xtb, NF, hin, NF, NF, 2 * NF, WcatT, NF, bcat,
                                         nullptr, hout, NF, 1, nullptr, 0, nullptr);
    // autoencoder (BN folded into weights/bias), bf16 activations
    gemm_wmma<<<gN1024, blk, 0, stream>>>(hout, NF, nullptr, 0, NF, NF, W1T, HIDP, bb1,
                                          nullptr, z1, HIDP, 0, nullptr, 0, nullptr);
    gemm_wmma<<<gN1024, blk, 0, stream>>>(z1, HIDP, nullptr, 0, HIDP, HIDP, W2T, HIDP, bb2,
                                          nullptr, z2, HIDP, 0, nullptr, 0, nullptr);
    const float* lossX = (t < TS - 1) ? (x + (size_t)(t + 1) * NF) : nullptr;
    gemm_wmma<<<gN512, blk, 0, stream>>>(z2, HIDP, nullptr, 0, HIDP, HIDP, W3T, NF, bb3,
                                         yb, ybb, NF, 0, lossX, TT * NF, lacc);
  }
  // after loop: hA holds h[:, -1] (t=63 wrote hA), yb/ybb hold y[:, -1]

  // ---- decoder rollout (16 steps), gather x_pred by (t-1) ----
  for (int s = 0; s < 16; ++s) {
    const __bf16* hin  = (s & 1) ? hB : hA;
    __bf16*       hout = (s & 1) ? hA : hB;
    gemm_wmma<<<gN512, blk, 0, stream>>>(ybb, NF, hin, NF, NF, 2 * NF, WcatT, NF, bcat,
                                         nullptr, hout, NF, 1, nullptr, 0, nullptr);
    gemm_wmma<<<gN1024, blk, 0, stream>>>(hout, NF, nullptr, 0, NF, NF, W1T, HIDP, bb1,
                                          nullptr, z1, HIDP, 0, nullptr, 0, nullptr);
    gemm_wmma<<<gN1024, blk, 0, stream>>>(z1, HIDP, nullptr, 0, HIDP, HIDP, W2T, HIDP, bb2,
                                          nullptr, z2, HIDP, 0, nullptr, 0, nullptr);
    gemm_wmma<<<gN512, blk, 0, stream>>>(z2, HIDP, nullptr, 0, HIDP, HIDP, W3T, NF, bb3,
                                         yb, ybb, NF, 0, nullptr, 0, nullptr);
    select_pred<<<BSZ, 256, 0, stream>>>(yb, tt, s, outp);
  }

  loss2_kernel<<<BSZ, 256, 0, stream>>>(outp, x, lacc);
  finalize<<<1, 1, 0, stream>>>(lacc, outp + (size_t)BSZ * NF);
}
